// CellForecaster_39625368273273
// MI455X (gfx1250) — compile-verified, matrix operand order checked by reading, not run on hardware
//
#include <hip/hip_runtime.h>
#include <hip/hip_bf16.h>

// ---------------------------------------------------------------------------
// LSTM forecaster for MI455X (gfx1250, wave32, WMMA).
// B=4096, T=512, H=128, IN=2, OUT=2, FUTURE=50.
// Persistent per-batch-tile kernel: 16 batch rows / block, 8 waves / block,
// wave w owns hidden cols [16w,16w+16). W_hh held in VGPRs as f16 WMMA B
// fragments; h exchanged through double-buffered LDS (one barrier/step);
// c + accumulators f32; activations via native gfx1250 v_tanh when available.
// ---------------------------------------------------------------------------

#define B_TOT   4096
#define T_SEQ   512
#define HIDDEN  128
#define IN_SZ   2
#define OUT_SZ  2
#define FUTURE  50

typedef _Float16 half_t;
typedef __attribute__((ext_vector_type(16))) _Float16 v16h;
typedef __attribute__((ext_vector_type(8)))  _Float16 v8h;
typedef __attribute__((ext_vector_type(8)))  float    v8f;

__device__ __forceinline__ float ftanh(float x) {
#if __has_builtin(__builtin_amdgcn_tanhf)
    return __builtin_amdgcn_tanhf(x);            // native v_tanh_f32 (gfx1250 TRANS)
#elif __has_builtin(__builtin_amdgcn_tanh_f32)
    return __builtin_amdgcn_tanh_f32(x);
#else
    float e = __expf(-2.0f * x);
#if __has_builtin(__builtin_amdgcn_rcpf)
    return (1.0f - e) * __builtin_amdgcn_rcpf(1.0f + e);
#else
    return (1.0f - e) / (1.0f + e);
#endif
#endif
}
__device__ __forceinline__ float fsig(float x) {
    // sigmoid(x) = 0.5*tanh(x/2) + 0.5  -> one TRANS op + one FMA
    return 0.5f * ftanh(0.5f * x) + 0.5f;
}

__global__ __launch_bounds__(256)
void lstm_forecast_kernel(const float* __restrict__ x,
                          const float* __restrict__ W_ih,
                          const float* __restrict__ W_hh,
                          const float* __restrict__ b_ih,
                          const float* __restrict__ b_hh,
                          const float* __restrict__ fc_w,
                          const float* __restrict__ fc_b,
                          float* __restrict__ out) {
    // Double-buffered LDS: h (f16, WMMA A source) and x/y staging.
    __shared__ alignas(16) half_t h_sh[2][16][HIDDEN];    // 8 KB
    __shared__ alignas(16) float  x_sh[2][16][IN_SZ];     // 256 B
    __shared__ alignas(16) float  fcw_sh[OUT_SZ][HIDDEN]; // 1 KB
    __shared__             float  fcb_sh[OUT_SZ];

    const int tid  = threadIdx.x;
    const int lane = tid & 31;
    const int wave = tid >> 5;
    const int hi   = lane >> 4;   // half-lane group: 0 or 1
    const int n    = lane & 15;   // column within 16-wide tile
    const int b0   = blockIdx.x * 16;
    const int hc0  = wave * 16;   // hidden columns owned by this wave

    // ---- one-time init -----------------------------------------------------
    for (int i = tid; i < 16 * HIDDEN; i += 256)
        (&h_sh[0][0][0])[i] = (half_t)0.0f;     // buffer 0 = h at t=0
    if (tid < OUT_SZ * HIDDEN)
        (&fcw_sh[0][0])[tid] = fc_w[tid];
    if (tid < OUT_SZ)
        fcb_sh[tid] = fc_b[tid];
    if (tid < 32) {                             // x for t = 0 -> buffer 0
        int m = tid >> 1, cc = tid & 1;
        x_sh[0][m][cc] = x[((size_t)(b0 + m) * T_SEQ + 0) * IN_SZ + cc];
    }

    // W_hh -> per-wave f16 WMMA B fragments (resident for all 562 steps).
    // B[k, nn] = W_hh[g*128 + hc0 + nn, kc*32 + k]; 16-bit fragment layout:
    //   element j<8  -> k = hi*8 + j
    //   element j>=8 -> k = 16 + hi*8 + (j-8)
    v16h Bw[4][4];
    float biasL[4], wi0[4], wi1[4];
#pragma unroll
    for (int g = 0; g < 4; ++g) {
        const int row = g * HIDDEN + hc0 + n;
        biasL[g] = b_ih[row] + b_hh[row];
        wi0[g]   = W_ih[row * IN_SZ + 0];
        wi1[g]   = W_ih[row * IN_SZ + 1];
#pragma unroll
        for (int kc = 0; kc < 4; ++kc) {
            const float* p = W_hh + (size_t)row * HIDDEN + kc * 32 + hi * 8;
            v16h bb;
#pragma unroll
            for (int j = 0; j < 8; ++j)  bb[j] = (half_t)p[j];
#pragma unroll
            for (int j = 8; j < 16; ++j) bb[j] = (half_t)p[j + 8];
            Bw[g][kc] = bb;
        }
    }

    v8f cst;                       // cell state c[m][hc0+n], f32, register-resident
#pragma unroll
    for (int v = 0; v < 8; ++v) cst[v] = 0.0f;

    __syncthreads();

    // ---- one LSTM cell step: reads h_sh[p]/x_sh[p], writes h_sh[p^1] -------
    auto cell_step = [&](int p) {
        // input term + biases into accumulators
        float x0[8], x1[8];
#pragma unroll
        for (int v = 0; v < 8; ++v) {
            int m = v + hi * 8;
            x0[v] = x_sh[p][m][0];
            x1[v] = x_sh[p][m][1];
        }
        v8f acc[4];
#pragma unroll
        for (int g = 0; g < 4; ++g)
#pragma unroll
            for (int v = 0; v < 8; ++v)
                acc[g][v] = biasL[g] + x0[v] * wi0[g] + x1[v] * wi1[g];

        // gates += h @ W_hh^T  (A shared across waves, B in registers)
#pragma unroll
        for (int kc = 0; kc < 4; ++kc) {
            union { v16h v; v8h h2[2]; } a;
            a.h2[0] = *(const v8h*)&h_sh[p][n][kc * 32 + hi * 8];
            a.h2[1] = *(const v8h*)&h_sh[p][n][kc * 32 + 16 + hi * 8];
#pragma unroll
            for (int g = 0; g < 4; ++g)
                acc[g] = __builtin_amdgcn_wmma_f32_16x16x32_f16(
                    false, a.v, false, Bw[g][kc], (short)0, acc[g], false, false);
        }

        // elementwise LSTM update; write h_new (f16) into the other buffer
#pragma unroll
        for (int v = 0; v < 8; ++v) {
            float is = fsig(acc[0][v]);
            float fs = fsig(acc[1][v]);
            float gt = ftanh(acc[2][v]);
            float os = fsig(acc[3][v]);
            float cn = fs * cst[v] + is * gt;
            cst[v] = cn;
            float hn = os * ftanh(cn);
            int m = v + hi * 8;
            h_sh[p ^ 1][m][hc0 + n] = (half_t)hn;
        }
    };

    // ---- encode: 512 steps, single barrier per step (double-buffered) ------
    int p = 0;
    for (int t = 0; t < T_SEQ; ++t) {
        cell_step(p);
        if (tid < 32) {
            int m = tid >> 1, cc = tid & 1;
            if (t + 1 < T_SEQ)
                x_sh[p ^ 1][m][cc] = x[((size_t)(b0 + m) * T_SEQ + (t + 1)) * IN_SZ + cc];
            if (t + 8 < T_SEQ)   // global_prefetch_b8; 192MB L2 keeps x resident
                __builtin_prefetch(&x[((size_t)(b0 + m) * T_SEQ + (t + 8)) * IN_SZ + cc], 0, 0);
        }
        __syncthreads();
        p ^= 1;
    }

    // ---- forecast: 50 autoregressive steps ---------------------------------
    for (int tf = 0; tf < FUTURE; ++tf) {
        if (wave == 0) {
            // y[m][o] = fc_b[o] + h[m,:] . fc_w[o,:]   (32 lanes, 32 outputs)
            int m = n, o = hi;
            float acc = fcb_sh[o];
#pragma unroll
            for (int k = 0; k < HIDDEN; k += 8) {
                v8h hv = *(const v8h*)&h_sh[p][m][k];
#pragma unroll
                for (int j = 0; j < 8; ++j)
                    acc += (float)hv[j] * fcw_sh[o][k + j];
            }
            out[(size_t)(b0 + m) * (FUTURE * OUT_SZ) + tf * OUT_SZ + o] = acc;
            x_sh[p][m][o] = acc;    // feed back as this step's cell input
        }
        __syncthreads();
        cell_step(p);
        __syncthreads();
        p ^= 1;
    }
}

extern "C" void kernel_launch(void* const* d_in, const int* in_sizes, int n_in,
                              void* d_out, int out_size, void* d_ws, size_t ws_size,
                              hipStream_t stream) {
    const float* x    = (const float*)d_in[0];
    const float* W_ih = (const float*)d_in[1];
    const float* W_hh = (const float*)d_in[2];
    const float* b_ih = (const float*)d_in[3];
    const float* b_hh = (const float*)d_in[4];
    const float* fc_w = (const float*)d_in[5];
    const float* fc_b = (const float*)d_in[6];
    float* out = (float*)d_out;
    (void)in_sizes; (void)n_in; (void)out_size; (void)d_ws; (void)ws_size;

    dim3 grid(B_TOT / 16);   // 256 independent batch tiles
    dim3 block(256);         // 8 wave32 per tile
    lstm_forecast_kernel<<<grid, block, 0, stream>>>(x, W_ih, W_hh, b_ih, b_hh,
                                                     fc_w, fc_b, out);
}
